// LayerNormGRUCell_64982855188492
// MI455X (gfx1250) — matrix operand order repaired and checked by reading, not compiled, not run on hardware
//
#include <hip/hip_runtime.h>
#include <hip/hip_bf16.h>

// ---------------------------------------------------------------------------
// LayerNorm GRU cell on MI455X (gfx1250, wave32, WMMA).
//   B = 8192, I = H = 1024.
//   Stage 1: f32 -> bf16 pack of x, h, and [W_irz;W_in], [W_hrz;W_hn].
//   Stage 2: bf16 WMMA GEMM  (8192 x 3072) = A(8192x1024) @ W^T(1024x3072)
//            f32 accum -> scratch (pre-LN activations), run for x and h side.
//            Workgroup tile 32M x 512N, per-wave register tile 32M x 64N
//            (8 WMMAs per K-step), A-block staged in LDS via the gfx1250
//            async global->LDS path (ASYNCcnt) when available.
//   Stage 3: per-row LayerNorm (segments [0,2048) and [2048,3072)) + GRU
//            pointwise combine -> out.
// ---------------------------------------------------------------------------

#define B_ROWS 8192
#define KDIM   1024
#define NDIM   3072   // 2048 (rz) + 1024 (n)

typedef __attribute__((ext_vector_type(16))) __bf16 v16bf;
typedef __attribute__((ext_vector_type(8)))  __bf16 v8bf;
typedef __attribute__((ext_vector_type(8)))  float  v8f;
typedef __attribute__((ext_vector_type(4)))  int    v4i;

#if defined(__has_builtin)
#if __has_builtin(__builtin_amdgcn_global_load_async_to_lds_b128) && \
    __has_builtin(__builtin_amdgcn_s_wait_asynccnt)
#define LNGRU_ASYNC_LDS 1
#endif
#endif

// ---------------------------------------------------------------- converters
__global__ __launch_bounds__(256) void lngru_cvt_bf16(const float* __restrict__ src,
                                                      __bf16* __restrict__ dst, int n) {
    int i = blockIdx.x * 256 + threadIdx.x;
    if (i < n) dst[i] = (__bf16)src[i];
}

__global__ __launch_bounds__(256) void lngru_copy_f32(const float* __restrict__ src,
                                                      float* __restrict__ dst, int n) {
    int i = blockIdx.x * 256 + threadIdx.x;
    if (i < n) dst[i] = src[i];
}

// ------------------------------------------------------------------ WMMA GEMM
// grid = (NDIM/512, B_ROWS/32), block = 256 (8 waves).
// Workgroup: rows [m0, m0+32), cols [512*bx, 512*bx+512).
// Wave w owns cols [512*bx + 64*w, +64): 2 M-tiles x 4 N-tiles in registers.
__global__ __launch_bounds__(256) void lngru_gemm_bf16(const __bf16* __restrict__ A,
                                                       const __bf16* __restrict__ W,
                                                       float* __restrict__ C) {
    __shared__ __bf16 As[32 * KDIM];   // 64 KB

    const int tid  = threadIdx.x;
    const int wave = tid >> 5;
    const int lane = tid & 31;
    const int m0   = blockIdx.y * 32;
    const int n0w  = blockIdx.x * 512 + wave * 64;

    // ---- Stage A rows m0..m0+31 (64 KB, contiguous in row-major A) into LDS.
#if defined(LNGRU_ASYNC_LDS)
    {
        // gfx1250 async copy: global -> LDS, tracked by ASYNCcnt (no VGPR hop).
        // Builtin signature (from clang): (v4i AS1*, v4i AS3*, i32 offset, i32 cpol)
        typedef __attribute__((address_space(1))) v4i* gv4i_t;
        typedef __attribute__((address_space(3))) v4i* lv4i_t;
        char* gsrc = (char*)(A + (size_t)m0 * KDIM);   // const dropped for builtin
        char* lbase = (char*)As;
        #pragma unroll
        for (int i = tid; i < (32 * KDIM * 2) / 16; i += 256) {
            __builtin_amdgcn_global_load_async_to_lds_b128(
                (gv4i_t)(gsrc + (size_t)i * 16), (lv4i_t)(lbase + (size_t)i * 16),
                /*offset=*/0, /*cpol=*/0);
        }
        __builtin_amdgcn_s_wait_asynccnt(0);
    }
#else
    {
        const uint4* gsrc = (const uint4*)(A + (size_t)m0 * KDIM);
        uint4* ldst = (uint4*)As;
        #pragma unroll
        for (int i = tid; i < (32 * KDIM) / 8; i += 256)
            ldst[i] = gsrc[i];
    }
#endif
    __syncthreads();

    const int lhalf = lane >> 4;   // 0: lanes 0-15, 1: lanes 16-31
    const int lm    = lane & 15;

    // B column n of the (K x N) operand == row (n) of W, contiguous over K.
    // Fragment layout (32x16 bf16): lanes 0-15 hold K=kc..kc+15 of column lm,
    // lanes 16-31 hold K=kc+16..kc+31 (one contiguous 32B run per lane).
    const __bf16* Wn[4];
    #pragma unroll
    for (int nt = 0; nt < 4; ++nt) {
        Wn[nt] = W + (size_t)(n0w + nt * 16 + lm) * KDIM + lhalf * 16;
        __builtin_prefetch(Wn[nt], 0, 1);            // global_prefetch_b8
    }

    v8f acc[2][4] = {};

    #pragma unroll 2
    for (int kc = 0; kc < KDIM; kc += 32) {
        // A fragments 16x32 bf16 (contiguous K, matching B), one per M-tile:
        //   lanes 0-15 : M=lm, K = kc+0..15
        //   lanes16-31 : M=lm, K = kc+16..31
        v16bf afr[2];
        #pragma unroll
        for (int mt = 0; mt < 2; ++mt) {
            const __bf16* ap = &As[(mt * 16 + lm) * KDIM + kc + lhalf * 16];
            v8bf lo = *(const v8bf*)ap;
            v8bf hi = *(const v8bf*)(ap + 8);
            #pragma unroll
            for (int i = 0; i < 8; ++i) { afr[mt][i] = lo[i]; afr[mt][i + 8] = hi[i]; }
        }

        v16bf bfr[4];
        #pragma unroll
        for (int nt = 0; nt < 4; ++nt)
            bfr[nt] = *(const v16bf*)(Wn[nt] + kc);

        #pragma unroll
        for (int nt = 0; nt < 4; ++nt)
            #pragma unroll
            for (int mt = 0; mt < 2; ++mt)
                acc[mt][nt] = __builtin_amdgcn_wmma_f32_16x16x32_bf16(
                    /*neg_a=*/false, afr[mt], /*neg_b=*/false, bfr[nt],
                    /*c_mod=*/(short)0, acc[mt][nt],
                    /*reuse_a=*/false, /*reuse_b=*/false);
    }

    // C tiles: VGPR r -> (M = r + 8*lhalf, N = lm), f32 row-major (B_ROWS x NDIM)
    #pragma unroll
    for (int mt = 0; mt < 2; ++mt) {
        #pragma unroll
        for (int nt = 0; nt < 4; ++nt) {
            float* Cp = C + (size_t)(m0 + mt * 16 + lhalf * 8) * NDIM
                          + (n0w + nt * 16 + lm);
            #pragma unroll
            for (int r = 0; r < 8; ++r)
                Cp[(size_t)r * NDIM] = acc[mt][nt][r];
        }
    }
}

// ----------------------------------------------------- LayerNorm + GRU combine
__device__ __forceinline__ float lngru_wsum(float v) {
    #pragma unroll
    for (int m = 16; m > 0; m >>= 1) v += __shfl_xor(v, m, 32);
    return v;
}
__device__ __forceinline__ float lngru_sigmoid(float t) {
    return 1.0f / (1.0f + __expf(-t));
}

// grid = B_ROWS, block = 256. One row per workgroup.
__global__ __launch_bounds__(256) void lngru_ln_combine(const float* __restrict__ Ax,
                                                        const float* __restrict__ Ah,
                                                        const float* __restrict__ bx,
                                                        const float* __restrict__ bh,
                                                        const float* __restrict__ hrow,
                                                        float* __restrict__ out) {
    __shared__ float vx[NDIM];
    __shared__ float vh[NDIM];
    __shared__ float red[8];

    const int tid = threadIdx.x;
    const size_t row = blockIdx.x;

    const float* axp = Ax + row * NDIM;
    const float* ahp = Ah + row * NDIM;
    for (int c = tid; c < NDIM; c += 256) {
        vx[c] = axp[c] + bx[c];
        vh[c] = ahp[c] + bh[c];
    }
    if (tid < 8) red[tid] = 0.0f;
    __syncthreads();

    // segment 1: cols [0, 2048)  (rz, LN over 2048)
    // segment 2: cols [2048, 3072) (n, LN over 1024)
    float s1x = 0.f, q1x = 0.f, s1h = 0.f, q1h = 0.f;
    float s2x = 0.f, q2x = 0.f, s2h = 0.f, q2h = 0.f;
    for (int c = tid; c < 2048; c += 256) {
        float a = vx[c]; s1x += a; q1x += a * a;
        float b = vh[c]; s1h += b; q1h += b * b;
    }
    for (int c = 2048 + tid; c < NDIM; c += 256) {
        float a = vx[c]; s2x += a; q2x += a * a;
        float b = vh[c]; s2h += b; q2h += b * b;
    }
    s1x = lngru_wsum(s1x); q1x = lngru_wsum(q1x);
    s1h = lngru_wsum(s1h); q1h = lngru_wsum(q1h);
    s2x = lngru_wsum(s2x); q2x = lngru_wsum(q2x);
    s2h = lngru_wsum(s2h); q2h = lngru_wsum(q2h);
    if ((tid & 31) == 0) {
        atomicAdd(&red[0], s1x); atomicAdd(&red[1], q1x);
        atomicAdd(&red[2], s1h); atomicAdd(&red[3], q1h);
        atomicAdd(&red[4], s2x); atomicAdd(&red[5], q2x);
        atomicAdd(&red[6], s2h); atomicAdd(&red[7], q2h);
    }
    __syncthreads();

    const float EPS = 1e-5f;
    float mu1x = red[0] * (1.0f / 2048.0f);
    float rs1x = rsqrtf(red[1] * (1.0f / 2048.0f) - mu1x * mu1x + EPS);
    float mu1h = red[2] * (1.0f / 2048.0f);
    float rs1h = rsqrtf(red[3] * (1.0f / 2048.0f) - mu1h * mu1h + EPS);
    float mu2x = red[4] * (1.0f / 1024.0f);
    float rs2x = rsqrtf(red[5] * (1.0f / 1024.0f) - mu2x * mu2x + EPS);
    float mu2h = red[6] * (1.0f / 1024.0f);
    float rs2h = rsqrtf(red[7] * (1.0f / 1024.0f) - mu2h * mu2h + EPS);

    const float* hp = hrow + row * 1024;
    float* op = out + row * 1024;
    for (int c = tid; c < 1024; c += 256) {
        float irz_r = (vx[c]          - mu1x) * rs1x;
        float hrz_r = (vh[c]          - mu1h) * rs1h;
        float irz_z = (vx[c + 1024]   - mu1x) * rs1x;
        float hrz_z = (vh[c + 1024]   - mu1h) * rs1h;
        float r = lngru_sigmoid(irz_r + hrz_r);
        float z = lngru_sigmoid(irz_z + hrz_z);
        float i_n = (vx[c + 2048] - mu2x) * rs2x;
        float h_n = (vh[c + 2048] - mu2h) * rs2h;
        float n = tanhf(i_n + r * h_n);
        float hv = hp[c];
        op[c] = (1.0f - z) * n + z * hv;
    }
}

// ---------------------------------------------------------------------- launch
extern "C" void kernel_launch(void* const* d_in, const int* in_sizes, int n_in,
                              void* d_out, int out_size, void* d_ws, size_t ws_size,
                              hipStream_t stream) {
    const float* x     = (const float*)d_in[0];
    const float* h     = (const float*)d_in[1];
    const float* W_irz = (const float*)d_in[2];
    const float* b_irz = (const float*)d_in[3];
    const float* W_hrz = (const float*)d_in[4];
    const float* b_hrz = (const float*)d_in[5];
    const float* W_in  = (const float*)d_in[6];
    const float* b_in  = (const float*)d_in[7];
    const float* W_hn  = (const float*)d_in[8];
    const float* b_hn  = (const float*)d_in[9];
    float* out = (float*)d_out;

    size_t off = 0;
    auto alloc = [&](size_t bytes) -> void* {
        void* p = (char*)d_ws + off;
        off = (off + bytes + 255) & ~(size_t)255;
        return p;
    };

    const size_t XH = (size_t)B_ROWS * KDIM;          // 8M elems
    const size_t WN = (size_t)NDIM * KDIM;            // 3M elems

    __bf16* Xb = (__bf16*)alloc(XH * sizeof(__bf16));
    __bf16* Hb = (__bf16*)alloc(XH * sizeof(__bf16));
    __bf16* Wx = (__bf16*)alloc(WN * sizeof(__bf16));
    __bf16* Wh = (__bf16*)alloc(WN * sizeof(__bf16));
    float*  bx = (float*)alloc(NDIM * sizeof(float));
    float*  bh = (float*)alloc(NDIM * sizeof(float));
    float*  Ax = (float*)alloc((size_t)B_ROWS * NDIM * sizeof(float));
    float*  Ah = (float*)alloc((size_t)B_ROWS * NDIM * sizeof(float));

    // Stage 1: pack/convert.
    auto blocks = [](size_t n) { return (unsigned)((n + 255) / 256); };
    lngru_cvt_bf16<<<blocks(XH), 256, 0, stream>>>(x, Xb, (int)XH);
    lngru_cvt_bf16<<<blocks(XH), 256, 0, stream>>>(h, Hb, (int)XH);
    lngru_cvt_bf16<<<blocks(2048 * KDIM), 256, 0, stream>>>(W_irz, Wx, 2048 * KDIM);
    lngru_cvt_bf16<<<blocks(1024 * KDIM), 256, 0, stream>>>(W_in, Wx + (size_t)2048 * KDIM, 1024 * KDIM);
    lngru_cvt_bf16<<<blocks(2048 * KDIM), 256, 0, stream>>>(W_hrz, Wh, 2048 * KDIM);
    lngru_cvt_bf16<<<blocks(1024 * KDIM), 256, 0, stream>>>(W_hn, Wh + (size_t)2048 * KDIM, 1024 * KDIM);
    lngru_copy_f32<<<blocks(2048), 256, 0, stream>>>(b_irz, bx, 2048);
    lngru_copy_f32<<<blocks(1024), 256, 0, stream>>>(b_in, bx + 2048, 1024);
    lngru_copy_f32<<<blocks(2048), 256, 0, stream>>>(b_hrz, bh, 2048);
    lngru_copy_f32<<<blocks(1024), 256, 0, stream>>>(b_hn, bh + 2048, 1024);

    // Stage 2: WMMA GEMMs (x-side, h-side).
    dim3 ggrid(NDIM / 512, B_ROWS / 32);
    lngru_gemm_bf16<<<ggrid, 256, 0, stream>>>(Xb, Wx, Ax);
    lngru_gemm_bf16<<<ggrid, 256, 0, stream>>>(Hb, Wh, Ah);

    // Stage 3: LayerNorm + GRU combine.
    lngru_ln_combine<<<B_ROWS, 256, 0, stream>>>(Ax, Ah, bx, bh, h, out);
}
